// SNN_41042707480695
// MI455X (gfx1250) — matrix operand order repaired and checked by reading, not compiled, not run on hardware
//
#include <hip/hip_runtime.h>
#include <hip/hip_bf16.h>

typedef __attribute__((ext_vector_type(16))) __bf16 v16bf;
typedef __attribute__((ext_vector_type(8)))  float  v8f;

// ---------------------------------------------------------------------------
// CDNA5 async global->LDS copy (16B), tracked by ASYNCcnt. No VGPR round-trip.
// lds_off = byte offset inside the workgroup LDS allocation (low 32 bits of
// the flat shared pointer per ISA §10.2 LDS aperture mapping).
// ---------------------------------------------------------------------------
__device__ __forceinline__ void async_ld16(unsigned lds_off, const void* gsrc) {
    asm volatile("global_load_async_to_lds_b128 %0, %1, off"
                 :: "v"(lds_off), "v"(gsrc)
                 : "memory");
}
__device__ __forceinline__ unsigned lds_addr(const void* p) {
    return (unsigned)(size_t)p;   // flat shared ptr low 32 bits == LDS offset
}

// ---------------------------------------------------------------------------
// fp32 -> bf16 staging, 8 elements/thread (two float4 in, one uint4 out)
// ---------------------------------------------------------------------------
__global__ void f32_to_bf16_kernel(const float* __restrict__ in,
                                   __hip_bfloat16* __restrict__ out, int n8) {
    int i = blockIdx.x * blockDim.x + threadIdx.x;
    if (i >= n8) return;
    const float4* p = reinterpret_cast<const float4*>(in) + (size_t)i * 2;
    float4 f0 = p[0], f1 = p[1];
    union { __hip_bfloat16 h[8]; uint4 u; } pk;
    pk.h[0] = __float2bfloat16(f0.x); pk.h[1] = __float2bfloat16(f0.y);
    pk.h[2] = __float2bfloat16(f0.z); pk.h[3] = __float2bfloat16(f0.w);
    pk.h[4] = __float2bfloat16(f1.x); pk.h[5] = __float2bfloat16(f1.y);
    pk.h[6] = __float2bfloat16(f1.z); pk.h[7] = __float2bfloat16(f1.w);
    reinterpret_cast<uint4*>(out)[i] = pk.u;
}

// ---------------------------------------------------------------------------
// C[M,N] = A[M,K](bf16) * W[N,K]^T(bf16) + bias[N]   (fp32 accumulate/output)
// Block tile 128x128, K-step 32, 256 threads = 8 waves (2 along M x 4 along N).
// Double-buffered LDS filled by async global->LDS; WMMA overlaps next fetch.
// Steady-state loop is branch-free: issue(k+1) -> wait 4 -> barrier -> wmma.
// ---------------------------------------------------------------------------
#define BM 128
#define BN 128
#define BK 32
#define LDP 40   // LDS row pitch in bf16 elements (80 bytes, 16B aligned)

__global__ __launch_bounds__(256)
void gemm_bf16_wmma(const __hip_bfloat16* __restrict__ A,
                    const __hip_bfloat16* __restrict__ W,
                    const float* __restrict__ bias,
                    float* __restrict__ C,
                    int M, int N, int K) {
    __shared__ __align__(16) unsigned short As[2][BM * LDP];
    __shared__ __align__(16) unsigned short Bs[2][BN * LDP];

    const int tid  = threadIdx.x;
    const int lane = tid & 31;
    const int wave = tid >> 5;      // 0..7
    const int mw   = wave & 1;      // wave M slot -> 64 rows
    const int nw   = wave >> 1;     // wave N slot -> 32 cols
    const int lr   = lane & 15;
    const int hs   = lane >> 4;     // lane-half select (ISA 7.12.2 layouts)

    const int blockM = blockIdx.y * BM;
    const int blockN = blockIdx.x * BN;

    // tile loader: each thread owns 16 bf16 (two async b128) per operand
    const int ldRow  = tid >> 1;    // 0..127
    const int ldHalf = tid & 1;     // K sub-chunk [0..15] or [16..31]

    // loop-carried global pointers (advance BK per issued tile)
    const __hip_bfloat16* gAcur = A + (size_t)(blockM + ldRow) * K + ldHalf * 16;
    const __hip_bfloat16* gBcur = W + (size_t)(blockN + ldRow) * K + ldHalf * 16;

    // LDS destinations: base of buffer 0, + constant stride for buffer 1
    const unsigned saBase = lds_addr(&As[0][ldRow * LDP + ldHalf * 16]);
    const unsigned sbBase = lds_addr(&Bs[0][ldRow * LDP + ldHalf * 16]);
    const unsigned bufStride = BM * LDP * 2;   // 10240 bytes per buffer

    auto issue_tile = [&](int buf) {
        const unsigned sa = saBase + buf * bufStride;
        const unsigned sb = sbBase + buf * bufStride;
        async_ld16(sa,      gAcur);      // cols +0..7
        async_ld16(sa + 16, gAcur + 8);  // cols +8..15
        async_ld16(sb,      gBcur);
        async_ld16(sb + 16, gBcur + 8);
        gAcur += BK;
        gBcur += BK;
    };

    v8f acc[4][2];
    const v8f vzero = {0.f, 0.f, 0.f, 0.f, 0.f, 0.f, 0.f, 0.f};
#pragma unroll
    for (int mi = 0; mi < 4; ++mi)
#pragma unroll
        for (int ni = 0; ni < 2; ++ni) acc[mi][ni] = vzero;

    union Frag { uint4 u[2]; v16bf v; };

    auto compute_tile = [&](int buf) {
        // ---- fragments: contiguous K per lane half (k = 16*hs + s), same for A and B ----
        Frag fa[4], fb[2];
#pragma unroll
        for (int ni = 0; ni < 2; ++ni) {
            // B 32x16: lane<16 -> col=lr, K 0..15 ; lane>=16 -> K 16..31
            const int n = nw * 32 + ni * 16 + lr;
            const char* base = reinterpret_cast<const char*>(Bs[buf]) + n * (LDP * 2);
            fb[ni].u[0] = *reinterpret_cast<const uint4*>(base + hs * 32);
            fb[ni].u[1] = *reinterpret_cast<const uint4*>(base + hs * 32 + 16);
        }
#pragma unroll
        for (int mi = 0; mi < 4; ++mi) {
            // A 16x32: lane<16 -> row=lr, K 0..15 ; lane>=16 -> K 16..31
            const int m = mw * 64 + mi * 16 + lr;
            const char* base = reinterpret_cast<const char*>(As[buf]) + m * (LDP * 2);
            fa[mi].u[0] = *reinterpret_cast<const uint4*>(base + hs * 32);
            fa[mi].u[1] = *reinterpret_cast<const uint4*>(base + hs * 32 + 16);
        }
#pragma unroll
        for (int mi = 0; mi < 4; ++mi)
#pragma unroll
            for (int ni = 0; ni < 2; ++ni)
                acc[mi][ni] = __builtin_amdgcn_wmma_f32_16x16x32_bf16(
                    false, fa[mi].v, false, fb[ni].v,
                    (short)0, acc[mi][ni], false, false);
    };

    const int kTiles = K / BK;
    issue_tile(0);                                   // ASYNCcnt = 4

    // steady state: branch-free pipeline
    for (int kt = 0; kt < kTiles - 1; ++kt) {
        const int buf = kt & 1;
        issue_tile(buf ^ 1);                         // ASYNCcnt = 8
        // async loads complete in order -> <=4 means current tile landed
        asm volatile("s_wait_asynccnt 0x4" ::: "memory");
        __syncthreads();                             // tile visible to all waves
        compute_tile(buf);
        __syncthreads();                             // safe to refill buf
    }

    // final (peeled) tile
    asm volatile("s_wait_asynccnt 0x0" ::: "memory");
    __syncthreads();
    compute_tile((kTiles - 1) & 1);

    // ---- epilogue: f32 16x16 C layout (VGPR r -> row r + 8*hs, col = lr) ----
#pragma unroll
    for (int mi = 0; mi < 4; ++mi) {
#pragma unroll
        for (int ni = 0; ni < 2; ++ni) {
            const int col  = blockN + nw * 32 + ni * 16 + lr;
            const float bv = bias[col];
            const int row0 = blockM + mw * 64 + mi * 16 + hs * 8;
#pragma unroll
            for (int r = 0; r < 8; ++r)
                C[(size_t)(row0 + r) * N + col] = acc[mi][ni][r] + bv;
        }
    }
}

// ---------------------------------------------------------------------------
// LIF recurrence over T for one (b,h) per thread. DECAY=0.9, THRESH=1.
// Layer 1 stores bf16 spikes (exact 0/1) for GEMM2; layer 2 stores only t=T-1.
// ---------------------------------------------------------------------------
template <bool STORE_ALL>
__global__ void lif_kernel(const float* __restrict__ iinj,
                           __hip_bfloat16* __restrict__ spikes,
                           float* __restrict__ last, int T, int H) {
    const int idx = blockIdx.x * blockDim.x + threadIdx.x;   // b*H + h
    const int b = idx / H;
    const int h = idx - b * H;
    const float* p = iinj + (size_t)b * T * H + h;
    __hip_bfloat16* sp = STORE_ALL ? (spikes + (size_t)b * T * H + h) : nullptr;
    float u = 0.f, s = 0.f;
#pragma unroll 4
    for (int t = 0; t < T; ++t) {
        const float it = p[(size_t)t * H];
        u = 0.9f * (u - s) + it;               // THRESH == 1
        s = (u > 1.0f) ? 1.0f : 0.0f;          // heaviside(u - 1)
        if (STORE_ALL) sp[(size_t)t * H] = __float2bfloat16(s);
    }
    if (!STORE_ALL) last[idx] = s;
}

// ---------------------------------------------------------------------------
// out[b,o] = sum_h hlast[b,h] * fcW[o,h] + fcb[o]      (32x256, trivial)
// ---------------------------------------------------------------------------
__global__ void fc_kernel(const float* __restrict__ hlast,
                          const float* __restrict__ fcW,
                          const float* __restrict__ fcb,
                          float* __restrict__ out, int Bn, int H, int O) {
    const int idx = blockIdx.x * blockDim.x + threadIdx.x;
    if (idx >= Bn * O) return;
    const int b = idx / O;
    const int o = idx - b * O;
    const float* hb = hlast + (size_t)b * H;
    const float* w  = fcW + (size_t)o * H;
    float acc = 0.f;
#pragma unroll 8
    for (int h = 0; h < H; ++h) acc = fmaf(hb[h], w[h], acc);
    out[idx] = acc + fcb[o];
}

// ---------------------------------------------------------------------------
extern "C" void kernel_launch(void* const* d_in, const int* in_sizes, int n_in,
                              void* d_out, int out_size, void* d_ws, size_t ws_size,
                              hipStream_t stream) {
    (void)in_sizes; (void)n_in; (void)out_size; (void)ws_size;
    const int B = 32, T = 1024, D = 512, H = 512, O = 256;

    const float* x   = (const float*)d_in[0];
    const float* W0  = (const float*)d_in[1];
    const float* b0  = (const float*)d_in[2];
    const float* W1  = (const float*)d_in[3];
    const float* b1  = (const float*)d_in[4];
    const float* fcW = (const float*)d_in[5];
    const float* fcb = (const float*)d_in[6];
    float* out = (float*)d_out;

    // workspace layout
    char* ws = (char*)d_ws;
    const size_t nBT  = (size_t)B * T;       // 32768 rows
    const size_t nBTH = nBT * H;             // 16,777,216 elements
    size_t off = 0;
    __hip_bfloat16* xb  = (__hip_bfloat16*)(ws + off); off += nBTH * 2;          // 33.5 MB
    __hip_bfloat16* h1b = (__hip_bfloat16*)(ws + off); off += nBTH * 2;          // 33.5 MB
    __hip_bfloat16* w0b = (__hip_bfloat16*)(ws + off); off += (size_t)H * D * 2; // 0.5 MB
    __hip_bfloat16* w1b = (__hip_bfloat16*)(ws + off); off += (size_t)H * H * 2; // 0.5 MB
    float* iinj   = (float*)(ws + off); off += nBTH * 4;                          // 67 MB (reused)
    float* h2last = (float*)(ws + off);                                           // 64 KB

    // 1) precision staging (vectorized 8 elems/thread)
    f32_to_bf16_kernel<<<(unsigned)((nBTH / 8 + 255) / 256), 256, 0, stream>>>(
        x, xb, (int)(nBTH / 8));
    f32_to_bf16_kernel<<<(H * D / 8 + 255) / 256, 256, 0, stream>>>(W0, w0b, H * D / 8);
    f32_to_bf16_kernel<<<(H * H / 8 + 255) / 256, 256, 0, stream>>>(W1, w1b, H * H / 8);

    const dim3 gg(H / BN, (unsigned)(nBT / BM));   // (4, 256)

    // 2) GEMM1: i_inj0 = x @ W0^T + b0
    gemm_bf16_wmma<<<gg, 256, 0, stream>>>(xb, w0b, b0, iinj, (int)nBT, H, D);
    // 3) LIF layer 1 -> bf16 spikes
    lif_kernel<true><<<(B * H) / 256, 256, 0, stream>>>(iinj, h1b, nullptr, T, H);
    // 4) GEMM2: i_inj1 = h1 @ W1^T + b1  (reuse iinj buffer)
    gemm_bf16_wmma<<<gg, 256, 0, stream>>>(h1b, w1b, b1, iinj, (int)nBT, H, H);
    // 5) LIF layer 2 -> final spike slice only
    lif_kernel<false><<<(B * H) / 256, 256, 0, stream>>>(iinj, nullptr, h2last, T, H);
    // 6) tiny FC head
    fc_kernel<<<(B * O) / 256, 256, 0, stream>>>(h2last, fcW, fcb, out, B, H, O);
}
